// EdgeGate_66683662238131
// MI455X (gfx1250) — matrix operand order repaired and checked by reading, not compiled, not run on hardware
//
#include <hip/hip_runtime.h>
#include <hip/hip_bf16.h>
#include <cstdint>

typedef __attribute__((ext_vector_type(16))) _Float16 v16h;
typedef __attribute__((ext_vector_type(8)))  float    v8f;

#define MOTIF_DIM 32
#define HIDDEN    64

// One wave32 handles 16 edges per tile:
//   A (16x32 f16)  : 4 fragments = [mu | mv | |mu-mv| | mu*mv] blocks of feats
//   B (32x16 f16)  : W1 chopped into 4 K-tiles x 4 N-tiles, resident in VGPRs
//   C (16x16 f32)  : h accumulators, 4 N-tiles
__global__ __launch_bounds__(256, 1) void edgegate_wmma_kernel(
    const float* __restrict__ motif_x,
    const long long* __restrict__ edge_index,
    const float* __restrict__ W1,
    const float* __restrict__ b1,
    const float* __restrict__ W2,
    const float* __restrict__ b2,
    float* __restrict__ out,
    int nEdges)
{
    const int lane  = threadIdx.x & 31;
    const int n0    = lane & 15;   // N column (B/C) and edge-row (A) within tile
    const int grp   = lane >> 4;   // half-wave: selects K sub-range
    const int kbase = grp * 8;     // A-fragment K pattern: {kb..kb+7, kb+16..kb+23}

    // ---- Stage W1 -> 16 f16 B fragments held in registers for the whole loop.
    // Dense 16-bit B 32x16 layout (matches A): lanes 0-15 hold K=0..7,16..23,
    // lanes 16-31 hold K=8..15,24..31; N = lane%16.
    v16h Bf[4][4];
    #pragma unroll
    for (int kt = 0; kt < 4; ++kt) {
        #pragma unroll
        for (int nt = 0; nt < 4; ++nt) {
            v16h bfrag;
            #pragma unroll
            for (int i = 0; i < 16; ++i) {
                const int k = kt * 32 + grp * 8 + (i < 8 ? i : i + 8); // K within 128
                bfrag[i] = (_Float16)W1[k * HIDDEN + nt * 16 + n0];
            }
            Bf[kt][nt] = bfrag;
        }
    }

    float b1v[4], w2v[4];
    #pragma unroll
    for (int c = 0; c < 4; ++c) {
        b1v[c] = b1[c * 16 + n0];
        w2v[c] = W2[c * 16 + n0];
    }
    const float b2s = b2[0];

    const long long* __restrict__ srcIdx = edge_index;
    const long long* __restrict__ dstIdx = edge_index + nEdges;

    const int nTiles = (nEdges + 15) >> 4;
    const int waveId = (int)((blockIdx.x * blockDim.x + threadIdx.x) >> 5);
    const int nWaves = (int)((gridDim.x * blockDim.x) >> 5);

    for (int tile = waveId; tile < nTiles; tile += nWaves) {
        const int base = tile << 4;
        const int e    = base + n0;                 // this lane's edge row
        const long long s = (e < nEdges) ? srcIdx[e] : 0;
        const long long d = (e < nEdges) ? dstIdx[e] : 0;

        const float* mu = motif_x + (size_t)s * MOTIF_DIM + kbase;
        const float* mv = motif_x + (size_t)d * MOTIF_DIM + kbase;

        // 8x global_load_b128 gathers (all hit L2: motif_x is 12.8MB << 192MB)
        const float4 u0 = *(const float4*)(mu);
        const float4 u1 = *(const float4*)(mu + 4);
        const float4 u2 = *(const float4*)(mu + 16);
        const float4 u3 = *(const float4*)(mu + 20);
        const float4 w0 = *(const float4*)(mv);
        const float4 w1 = *(const float4*)(mv + 4);
        const float4 w2 = *(const float4*)(mv + 16);
        const float4 w3 = *(const float4*)(mv + 20);

        const float uu[16] = {u0.x,u0.y,u0.z,u0.w, u1.x,u1.y,u1.z,u1.w,
                              u2.x,u2.y,u2.z,u2.w, u3.x,u3.y,u3.z,u3.w};
        const float vv[16] = {w0.x,w0.y,w0.z,w0.w, w1.x,w1.y,w1.z,w1.w,
                              w2.x,w2.y,w2.z,w2.w, w3.x,w3.y,w3.z,w3.w};

        v16h A0, A1, A2, A3;
        #pragma unroll
        for (int i = 0; i < 16; ++i) {
            const float du = uu[i], dv = vv[i];
            A0[i] = (_Float16)du;                    // mu        -> K   0..31
            A1[i] = (_Float16)dv;                    // mv        -> K  32..63
            A2[i] = (_Float16)fabsf(du - dv);        // |mu-mv|   -> K  64..95
            A3[i] = (_Float16)(du * dv);             // mu*mv     -> K 96..127
        }

        // h = feats @ W1 : 4 N-tiles, each a chain of 4 K-step WMMAs
        v8f C[4];
        #pragma unroll
        for (int nt = 0; nt < 4; ++nt) {
            v8f acc = {};
            acc = __builtin_amdgcn_wmma_f32_16x16x32_f16(false, A0, false, Bf[0][nt], (short)0, acc, false, false);
            acc = __builtin_amdgcn_wmma_f32_16x16x32_f16(false, A1, false, Bf[1][nt], (short)0, acc, false, false);
            acc = __builtin_amdgcn_wmma_f32_16x16x32_f16(false, A2, false, Bf[2][nt], (short)0, acc, false, false);
            acc = __builtin_amdgcn_wmma_f32_16x16x32_f16(false, A3, false, Bf[3][nt], (short)0, acc, false, false);
            C[nt] = acc;
        }

        // Layer 2: g = sigmoid(relu(h + b1) @ W2 + b2)
        // C layout: VGPR r, lanes 0-15 -> row r, lanes 16-31 -> row r+8; N = lane%16.
        float red[8];
        #pragma unroll
        for (int r = 0; r < 8; ++r) red[r] = 0.0f;
        #pragma unroll
        for (int c = 0; c < 4; ++c) {
            #pragma unroll
            for (int r = 0; r < 8; ++r) {
                float h = C[c][r] + b1v[c];
                h = fmaxf(h, 0.0f);
                red[r] = fmaf(h, w2v[c], red[r]);
            }
        }
        // XOR-tree reduction over the 16 lanes of each half-wave (masks < 16
        // never cross the half boundary).
        #pragma unroll
        for (int m = 8; m >= 1; m >>= 1) {
            #pragma unroll
            for (int r = 0; r < 8; ++r)
                red[r] += __shfl_xor(red[r], m, 32);
        }

        if (n0 == 0) {                  // lane 0 -> rows 0..7, lane 16 -> rows 8..15
            const int rb = base + grp * 8;
            #pragma unroll
            for (int r = 0; r < 8; ++r) {
                const int eo = rb + r;
                if (eo < nEdges) {
                    float g = 1.0f / (1.0f + __expf(-(red[r] + b2s)));
                    out[eo] = fminf(fmaxf(g, 0.0f), 1.0f);
                }
            }
        }
    }
}

extern "C" void kernel_launch(void* const* d_in, const int* in_sizes, int n_in,
                              void* d_out, int out_size, void* d_ws, size_t ws_size,
                              hipStream_t stream) {
    const float*     motif_x    = (const float*)d_in[0];
    const long long* edge_index = (const long long*)d_in[1];   // int64 in reference
    const float*     W1         = (const float*)d_in[2];
    const float*     b1         = (const float*)d_in[3];
    const float*     W2         = (const float*)d_in[4];
    const float*     b2         = (const float*)d_in[5];
    float*           out        = (float*)d_out;

    const int nEdges = in_sizes[1] / 2;          // edge_index is (2, E)

    const int threads       = 256;               // 8 waves / block
    const int wavesPerBlock = threads / 32;
    int nTiles = (nEdges + 15) / 16;
    int blocks = (nTiles + wavesPerBlock - 1) / wavesPerBlock;
    if (blocks > 512) blocks = 512;              // grid-stride: ~24 tiles/wave
    if (blocks < 1)   blocks = 1;

    edgegate_wmma_kernel<<<blocks, threads, 0, stream>>>(
        motif_x, edge_index, W1, b1, W2, b2, out, nEdges);
}
